// Model_87428354277646
// MI455X (gfx1250) — matrix operand order repaired and checked by reading, not compiled, not run on hardware
//
#include <hip/hip_runtime.h>

// Problem constants (from reference): B=4096, N=50, D=128, R=3, K=128
#define BATCH 4096
#define NTRI  50
#define DIM   128
#define TWOD  256
#define NREL  3
#define KDIM  128

typedef __attribute__((ext_vector_type(16))) __bf16 v16bf;
typedef __attribute__((ext_vector_type(8)))  float  v8f;

// ---- bf16 helpers via bit manipulation (no reliance on native cvt codegen) ----
__device__ __forceinline__ __bf16 bf16_trunc(float x) {
    unsigned short hs = (unsigned short)(__builtin_bit_cast(unsigned, x) >> 16);
    return __builtin_bit_cast(__bf16, hs);
}
__device__ __forceinline__ float bf16_to_f32(__bf16 h) {
    unsigned u = ((unsigned)__builtin_bit_cast(unsigned short, h)) << 16;
    return __builtin_bit_cast(float, u);
}
// Split fp32 into hi+lo bf16 pair: x ~= hi + lo, ~16 effective mantissa bits.
__device__ __forceinline__ void split_f32(float x, __bf16& hi, __bf16& lo) {
    hi = bf16_trunc(x);
    lo = bf16_trunc(x - bf16_to_f32(hi));
}

#define WMMA_BF16(A, B, C) \
    __builtin_amdgcn_wmma_f32_16x16x32_bf16(false, (A), false, (B), (short)0, (C), false, false)

__global__ __launch_bounds__(256) void transr_fused_kernel(
    const float* __restrict__ u_emb,   // [B, D]
    const float* __restrict__ i_emb,   // [B, D]
    const float* __restrict__ a_emb,   // [B, N, D]
    const float* __restrict__ o_emb,   // [B, N, D]
    const float* __restrict__ w_aor,   // [R, 2D, K]
    const float* __restrict__ w_uir,   // [R, 2D, K]
    const float* __restrict__ r_param, // [R, K]
    const int*   __restrict__ s,       // [B, N]
    float*       __restrict__ out)     // [B, N]
{
    // LDS: 8K + 8K + 24K + 16K + 3.2K ~= 59 KB
    __shared__ __bf16 ui_hi[16][TWOD];
    __shared__ __bf16 ui_lo[16][TWOD];
    __shared__ float  ui_all[16][NREL][KDIM];      // projected ui (+ r_param), fp32
    __shared__ __align__(16) float v_r[16][TWOD];  // W_aor[r] @ ui_all[.,r,:]
    __shared__ int    s_l[16 * NTRI];

    const int tid  = threadIdx.x;
    const int lane = tid & 31;
    const int wave = tid >> 5;
    const int b0   = blockIdx.x * 16;

    const int mrow = lane & 15;   // fragment row / column id within 16
    const int hb   = lane >> 4;   // half-of-wave selector

    // ---------------- Phase 0: stage ui_in = concat(u,i) as hi/lo bf16; stage s --------
    for (int i = tid; i < 16 * TWOD; i += 256) {
        const int row = i >> 8;
        const int d   = i & 255;
        const int b   = b0 + row;
        float x = (d < DIM) ? u_emb[(size_t)b * DIM + d]
                            : i_emb[(size_t)b * DIM + (d - DIM)];
        __bf16 h, l; split_f32(x, h, l);
        ui_hi[row][d] = h;
        ui_lo[row][d] = l;
    }
    for (int i = tid; i < 16 * NTRI; i += 256) {
        const int row = i / NTRI;
        const int n   = i - row * NTRI;
        s_l[i] = s[(size_t)(b0 + row) * NTRI + n];
    }
    __syncthreads();

    // ---------------- Phase 1: ui_all[row][r][k] = ui_in @ w_uir[r] + r_param[r] -------
    // 24 output tiles (3 relations x 8 k-tiles); 8 waves -> 3 tiles each (uniform).
    for (int t = wave; t < NREL * 8; t += 8) {
        const int r    = t >> 3;
        const int nt   = t & 7;
        const int ncol = nt * 16 + mrow;   // k column
        v8f acc = {0.f, 0.f, 0.f, 0.f, 0.f, 0.f, 0.f, 0.f};
        for (int kc = 0; kc < TWOD; kc += 32) {   // reduce over d
            v16bf a_hi, a_lo, b_hi, b_lo;
#pragma unroll
            for (int e = 0; e < 16; ++e) {        // A: 16x32, 16-bit layout
                const int kk = kc + e + 16 * hb;
                a_hi[e] = ui_hi[mrow][kk];
                a_lo[e] = ui_lo[mrow][kk];
            }
#pragma unroll
            for (int e = 0; e < 16; ++e) {        // B: 32x16, B[K=d][N=k] = w_uir[r][d][k]
                const int kd = kc + e + 16 * hb;
                float w = w_uir[((size_t)r * TWOD + kd) * KDIM + ncol];
                __bf16 h, l; split_f32(w, h, l);
                b_hi[e] = h; b_lo[e] = l;
            }
            acc = WMMA_BF16(a_hi, b_hi, acc);
            acc = WMMA_BF16(a_hi, b_lo, acc);
            acc = WMMA_BF16(a_lo, b_hi, acc);
        }
#pragma unroll
        for (int v = 0; v < 8; ++v) {             // C/D: M = v + 8*hb, N = mrow group
            ui_all[v + 8 * hb][r][ncol] = acc[v] + r_param[r * KDIM + ncol];
        }
    }
    __syncthreads();

    // ---------------- Phases 2+3 per relation ------------------------------------------
    for (int r = 0; r < NREL; ++r) {
        // Phase 2: v_r[row][d] = sum_k ui_all[row][r][k] * w_aor[r][d][k]
        // 16 output d-tiles; 8 waves -> 2 each (uniform).
        for (int t = wave; t < 16; t += 8) {
            const int ncol = t * 16 + mrow;   // d column
            v8f acc = {0.f, 0.f, 0.f, 0.f, 0.f, 0.f, 0.f, 0.f};
            for (int kc = 0; kc < KDIM; kc += 32) {   // reduce over k
                v16bf a_hi, a_lo, b_hi, b_lo;
#pragma unroll
                for (int e = 0; e < 16; ++e) {
                    const int kk = kc + e + 16 * hb;
                    float x = ui_all[mrow][r][kk];
                    __bf16 h, l; split_f32(x, h, l);
                    a_hi[e] = h; a_lo[e] = l;
                }
#pragma unroll
                for (int e = 0; e < 16; ++e) {    // B[K=k][N=d] = w_aor[r][d][k] (contig in k)
                    const int kd = kc + e + 16 * hb;
                    float w = w_aor[((size_t)r * TWOD + ncol) * KDIM + kd];
                    __bf16 h, l; split_f32(w, h, l);
                    b_hi[e] = h; b_lo[e] = l;
                }
                acc = WMMA_BF16(a_hi, b_hi, acc);
                acc = WMMA_BF16(a_hi, b_lo, acc);
                acc = WMMA_BF16(a_lo, b_hi, acc);
            }
#pragma unroll
            for (int v = 0; v < 8; ++v) v_r[v + 8 * hb][ncol] = acc[v];
        }
        __syncthreads();

        // Phase 3: streaming dot pass (memory-bound: 210 MB of a/o embeddings total).
        // One (row,n) pair per wave per iteration; s-test is wave-uniform.
        for (int idx = wave; idx < 16 * NTRI; idx += 8) {
            if (s_l[idx] != r) continue;
            const int row = idx / NTRI;
            const int n   = idx - row * NTRI;
            const size_t rowoff = ((size_t)(b0 + row) * NTRI + n) * DIM;
            const float4 a4 = ((const float4*)(a_emb + rowoff))[lane];
            const float4 o4 = ((const float4*)(o_emb + rowoff))[lane];
            const float4 va = *(const float4*)&v_r[row][lane * 4];
            const float4 vo = *(const float4*)&v_r[row][DIM + lane * 4];
            float sum = a4.x * va.x + a4.y * va.y + a4.z * va.z + a4.w * va.w
                      + o4.x * vo.x + o4.y * vo.y + o4.z * vo.z + o4.w * vo.w;
#pragma unroll
            for (int off = 16; off > 0; off >>= 1)
                sum += __shfl_xor(sum, off, 32);
            if (lane == 0) out[(size_t)(b0 + row) * NTRI + n] = sum;
        }
        __syncthreads();  // v_r reused next relation
    }
}

extern "C" void kernel_launch(void* const* d_in, const int* in_sizes, int n_in,
                              void* d_out, int out_size, void* d_ws, size_t ws_size,
                              hipStream_t stream) {
    (void)in_sizes; (void)n_in; (void)out_size; (void)d_ws; (void)ws_size;
    const float* u_emb   = (const float*)d_in[0];
    const float* i_emb   = (const float*)d_in[1];
    const float* a_emb   = (const float*)d_in[2];
    const float* o_emb   = (const float*)d_in[3];
    const float* w_aor   = (const float*)d_in[4];
    const float* w_uir   = (const float*)d_in[5];
    const float* r_param = (const float*)d_in[6];
    const int*   s       = (const int*)d_in[7];
    float*       out     = (float*)d_out;

    transr_fused_kernel<<<BATCH / 16, 256, 0, stream>>>(
        u_emb, i_emb, a_emb, o_emb, w_aor, w_uir, r_param, s, out);
}
